// Net_28973849379283
// MI455X (gfx1250) — matrix operand order repaired and checked by reading, not compiled, not run on hardware
//
#include <hip/hip_runtime.h>
#include <hip/hip_bf16.h>

typedef __attribute__((ext_vector_type(16))) _Float16 v16h;
typedef __attribute__((ext_vector_type(8)))  float    v8f;
typedef __attribute__((ext_vector_type(4)))  unsigned int u32x4;
typedef __attribute__((ext_vector_type(8)))  int      i32x8;
typedef __attribute__((ext_vector_type(4)))  int      i32x4;

#define WDIM 128
#define NPIX (WDIM*WDIM)          // 16384
#define NBATCH 8
#define NSEQ 16
#define BSEQ (NBATCH*NSEQ)        // 128
#define NVAL (BSEQ*NPIX)          // 2097152
#define NBINS 257
#define IDIM 5
#define EPSB 1e-5f
#define PI_F 3.14159265358979323846f

// padded channel-last f16 activation tensor: [img][PADH][PADW][32]
#define PADH 134                  // 3 + 128 + 3
#define PADW 136                  // 3 + 128 + 5  (keeps 64B-aligned tile starts)
#define PADPIX (PADH*PADW)        // 18224
#define PADN (NBATCH*PADPIX)      // 145792
#define ROWH (PADW*32)            // 4352 halves per padded row
#define IMGH ((size_t)PADH*ROWH)  // halves per padded image

// ---------------------------------------------------------------- utilities

__global__ __launch_bounds__(256) void k_zero(float* p, int n) {
    int gid = blockIdx.x * 256 + threadIdx.x;
    if (gid < n) p[gid] = 0.0f;
}

__global__ __launch_bounds__(256) void k_beam(const float* __restrict__ beam,
                                              float* __restrict__ invb) {
    __shared__ float red[256];
    int bs = blockIdx.x;
    float s = 0.0f;
    for (int i = threadIdx.x; i < NBINS; i += 256) s += beam[bs * NBINS + i];
    red[threadIdx.x] = s; __syncthreads();
    for (int st = 128; st > 0; st >>= 1) {
        if (threadIdx.x < st) red[threadIdx.x] += red[threadIdx.x + st];
        __syncthreads();
    }
    if (threadIdx.x == 0) invb[bs] = 1.0f / red[0];
}

__global__ __launch_bounds__(256) void k_vals(const float* __restrict__ pos,
                                              const float* __restrict__ beam,
                                              const float* __restrict__ invb,
                                              float* __restrict__ vals,
                                              float* __restrict__ vstats) {
    __shared__ float r1[256], r2[256];
    int gid = blockIdx.x * 256 + threadIdx.x;          // < NVAL
    int bs = gid >> 14, pix = gid & (NPIX - 1);
    int i = pix >> 7, j = pix & (WDIM - 1);
    float p0 = pos[bs * 2 + 0], p1 = pos[bs * 2 + 1];
    float th = atan2f((float)j - p1, (float)i - p0);
    int idx = (int)((th + PI_F) * ((float)NBINS / (2.0f * PI_F)));
    idx = idx < 0 ? 0 : (idx > NBINS - 1 ? NBINS - 1 : idx);
    float v = beam[bs * NBINS + idx] * invb[bs];
    vals[gid] = v;
    r1[threadIdx.x] = v; r2[threadIdx.x] = v * v; __syncthreads();
    for (int st = 128; st > 0; st >>= 1) {
        if (threadIdx.x < st) { r1[threadIdx.x] += r1[threadIdx.x + st];
                                r2[threadIdx.x] += r2[threadIdx.x + st]; }
        __syncthreads();
    }
    if (threadIdx.x == 0) { atomicAdd(&vstats[0], r1[0]); atomicAdd(&vstats[1], r2[0]); }
}

// bn1 affine fold: bn1(conv1(v)) = alpha*v + beta (conv1 is 1x1, 1 in-channel)
__global__ void k_aff1(const float* __restrict__ vstats,
                       const float* __restrict__ w1, const float* __restrict__ b1,
                       const float* __restrict__ g1, const float* __restrict__ bb1,
                       float* __restrict__ alpha, float* __restrict__ beta) {
    int c = threadIdx.x; if (c >= 32) return;
    float N = (float)NVAL;
    float mv = vstats[0] / N;
    float vv = vstats[1] / N - mv * mv;
    float w = w1[c];
    float inv = rsqrtf(w * w * vv + EPSB);
    float gi = g1[c] * inv;
    alpha[c] = gi * w;
    beta[c]  = -w * mv * gi + bb1[c];
    (void)b1;
}

__global__ __launch_bounds__(256) void k_zstats(const float* __restrict__ vals,
                                                const float* __restrict__ alpha,
                                                const float* __restrict__ beta,
                                                const float* __restrict__ w6,
                                                const float* __restrict__ b6,
                                                float* __restrict__ zstats) {
    __shared__ float sa[32], sb[32], sw[IDIM * 32], sbb[IDIM], ssum[IDIM], ssq[IDIM];
    int t = threadIdx.x;
    if (t < 32) { sa[t] = alpha[t]; sb[t] = beta[t]; }
    if (t < IDIM * 32) sw[t] = w6[t];
    if (t < IDIM) { sbb[t] = b6[t]; ssum[t] = 0.0f; ssq[t] = 0.0f; }
    __syncthreads();
    float v = vals[blockIdx.x * 256 + t];
    float z[IDIM];
    #pragma unroll
    for (int k = 0; k < IDIM; k++) z[k] = sbb[k];
    for (int c = 0; c < 32; c++) {
        float r = fmaxf(fmaf(sa[c], v, sb[c]), 0.0f);
        #pragma unroll
        for (int k = 0; k < IDIM; k++) z[k] = fmaf(sw[k * 32 + c], r, z[k]);
    }
    #pragma unroll
    for (int k = 0; k < IDIM; k++) { atomicAdd(&ssum[k], z[k]); atomicAdd(&ssq[k], z[k] * z[k]); }
    __syncthreads();
    if (t < IDIM) { atomicAdd(&zstats[t], ssum[t]); atomicAdd(&zstats[IDIM + t], ssq[t]); }
}

__global__ void k_aff6(const float* __restrict__ zstats,
                       const float* __restrict__ g6, const float* __restrict__ bb6,
                       float* __restrict__ a6, float* __restrict__ c6) {
    int k = threadIdx.x; if (k >= IDIM) return;
    float N = (float)NVAL;
    float m = zstats[k] / N;
    float var = zstats[IDIM + k] / N - m * m;
    float inv = rsqrtf(var + EPSB);
    a6[k] = g6[k] * inv;
    c6[k] = bb6[k] - m * g6[k] * inv;
}

__global__ __launch_bounds__(256) void k_xs(const float* __restrict__ vals,
                                            const float* __restrict__ alpha,
                                            const float* __restrict__ beta,
                                            const float* __restrict__ w6,
                                            const float* __restrict__ b6,
                                            const float* __restrict__ a6,
                                            const float* __restrict__ c6,
                                            const float* __restrict__ w7,
                                            const float* __restrict__ b7,
                                            float* __restrict__ xs,
                                            float* __restrict__ dsum) {
    __shared__ float sa[32], sb[32], sw[IDIM * 32], sbb[IDIM], sa6[IDIM], sc6[IDIM], sw7[IDIM];
    __shared__ float red[256];
    int t = threadIdx.x;
    if (t < 32) { sa[t] = alpha[t]; sb[t] = beta[t]; }
    if (t < IDIM * 32) sw[t] = w6[t];
    if (t < IDIM) { sbb[t] = b6[t]; sa6[t] = a6[t]; sc6[t] = c6[t]; sw7[t] = w7[t]; }
    __syncthreads();
    int gid = blockIdx.x * 256 + t;
    float v = vals[gid];
    float z[IDIM];
    #pragma unroll
    for (int k = 0; k < IDIM; k++) z[k] = sbb[k];
    for (int c = 0; c < 32; c++) {
        float r = fmaxf(fmaf(sa[c], v, sb[c]), 0.0f);
        #pragma unroll
        for (int k = 0; k < IDIM; k++) z[k] = fmaf(sw[k * 32 + c], r, z[k]);
    }
    float xv = b7[0];
    #pragma unroll
    for (int k = 0; k < IDIM; k++)
        xv = fmaf(sw7[k], fmaxf(fmaf(sa6[k], z[k], sc6[k]), 0.0f), xv);
    xs[gid] = xv;
    red[t] = xv; __syncthreads();
    for (int st = 128; st > 0; st >>= 1) {
        if (t < st) red[t] += red[t + st];
        __syncthreads();
    }
    if (t == 0) atomicAdd(&dsum[gid >> 14], red[0]);
}

__global__ __launch_bounds__(256) void k_dets(const float* __restrict__ xs,
                                              const float* __restrict__ dsum,
                                              float* __restrict__ out) {
    int gid = blockIdx.x * 256 + threadIdx.x;
    out[gid] = xs[gid] / dsum[gid >> 14];
}

// build padded channel-last f16 feature tensor: ch 0..15 = xs[s], 16/17/18 =
// max/min/mean over S, 19..31 = zero-pad; border pixels all-zero.
__global__ __launch_bounds__(256) void k_feat_pad(const float* __restrict__ xs,
                                                  _Float16* __restrict__ pf) {
    int gid = blockIdx.x * 256 + threadIdx.x;
    if (gid >= PADN) return;
    int img = gid / PADPIX, r = gid % PADPIX;
    int yp = r / PADW, xp = r % PADW;
    _Float16* dst = pf + (size_t)gid * 32;
    int y = yp - 3, x = xp - 3;
    if (y < 0 || y >= WDIM || x < 0 || x >= WDIM) {
        #pragma unroll
        for (int c = 0; c < 32; c++) dst[c] = (_Float16)0.0f;
        return;
    }
    int pix = y * WDIM + x;
    float mx = -3.4e38f, mn = 3.4e38f, sm = 0.0f;
    #pragma unroll
    for (int s = 0; s < NSEQ; s++) {
        float v = xs[(size_t)(img * NSEQ + s) * NPIX + pix];
        dst[s] = (_Float16)v;
        mx = fmaxf(mx, v); mn = fminf(mn, v); sm += v;
    }
    dst[16] = (_Float16)mx;
    dst[17] = (_Float16)mn;
    dst[18] = (_Float16)(sm * (1.0f / 16.0f));
    #pragma unroll
    for (int c = 19; c < 32; c++) dst[c] = (_Float16)0.0f;
}

// BN-affine + ReLU + f16 convert + pad: raw f32 [img][32][W][W] -> padded f16
__global__ __launch_bounds__(256) void k_xpad(const float* __restrict__ raw,
                                              const float* __restrict__ sc,
                                              const float* __restrict__ sh,
                                              _Float16* __restrict__ pf) {
    int gid = blockIdx.x * 256 + threadIdx.x;
    if (gid >= PADN) return;
    int img = gid / PADPIX, r = gid % PADPIX;
    int yp = r / PADW, xp = r % PADW;
    _Float16* dst = pf + (size_t)gid * 32;
    int y = yp - 3, x = xp - 3;
    if (y < 0 || y >= WDIM || x < 0 || x >= WDIM) {
        #pragma unroll
        for (int c = 0; c < 32; c++) dst[c] = (_Float16)0.0f;
        return;
    }
    const float* src = raw + ((size_t)img * 32 * NPIX) + y * WDIM + x;
    #pragma unroll
    for (int c = 0; c < 32; c++)
        dst[c] = (_Float16)fmaxf(fmaf(sc[c], src[(size_t)c * NPIX], sh[c]), 0.0f);
}

// pack conv weights (Cout, CinReal, 7,7) f32 -> f16 [tap][cout32][cin32(padded)]
__global__ __launch_bounds__(256) void k_prepw(const float* __restrict__ wsrc,
                                               int cin_real,
                                               _Float16* __restrict__ wdst) {
    int gid = blockIdx.x * 256 + threadIdx.x;
    if (gid >= 49 * 32 * 32) return;
    int ci = gid & 31, co = (gid >> 5) & 31, t = gid >> 10;
    float v = 0.0f;
    if (ci < cin_real) v = wsrc[((size_t)co * cin_real + ci) * 49 + t];
    wdst[gid] = (_Float16)v;
}

// ---------------------------------------------- TDM tile fetch (gfx1250)
// 2-D descriptor: 10 lines of 704 f16 elements (22 cols x 32 ch), line stride
// ROWH elements.  Tensor dims set large so nothing clips (tile is in-bounds
// by construction of the padded tensor).  D# packing per CDNA5 ISA ch.8.
__device__ inline void tdm_load_tile(unsigned lds_addr, unsigned long long ga) {
    u32x4 g0 = { 1u,                                   // count=1, user mode
                 lds_addr,                             // LDS byte address
                 (unsigned)ga,                         // global_addr[31:0]
                 (unsigned)(ga >> 32) | (2u << 30) };  // global_addr[56:32] | type=2
    i32x8 g1 = { 0x00010000,                           // wg_mask=0, data_size=1 (2B)
                 0,                                    // no atomic barrier
                 0x4000,                               // tensor_dim0 hi16 (=0x4000xxxx big)
                 (int)(0x4000u | (704u << 16)),        // tensor_dim1 hi16 | tile_dim0=704
                 10,                                   // tile_dim1=10, tile_dim2=0
                 (int)ROWH,                            // tensor_dim0_stride[31:0]
                 0, 0 };
    i32x4 gz = { 0, 0, 0, 0 };
#if __clang_major__ >= 23
    i32x8 gz8 = { 0, 0, 0, 0, 0, 0, 0, 0 };
    __builtin_amdgcn_tensor_load_to_lds(g0, g1, gz, gz, gz8, 0);
#else
    __builtin_amdgcn_tensor_load_to_lds(g0, g1, gz, gz, 0);
#endif
}

// ----------------------------------------------------- 7x7 conv via WMMA
// input: padded channel-last f16 tensor; tile DMA'd by the Tensor Data Mover.
// 256 thr = 8 waves; tile = 4 rows x 16 cols x 32 cout; K = 49 taps x 32 cin.
__global__ __launch_bounds__(256) void k_conv7(const _Float16* __restrict__ pin,
                                               const _Float16* __restrict__ wt,
                                               const float* __restrict__ bias,
                                               float* __restrict__ out,
                                               float* __restrict__ stats) {
    __shared__ _Float16 tin[10 * 22 * 32];           // 14080 B, matches TDM tile
    __shared__ float sstat[64];
    int tid = threadIdx.x;
    int img = blockIdx.z;
    int x0 = blockIdx.x * 16, y0 = blockIdx.y * 4;

    if (tid < 32) {                                  // one wave issues the DMA
        size_t off = (size_t)img * IMGH + (size_t)y0 * ROWH + (size_t)x0 * 32;
        unsigned long long ga = (unsigned long long)(uintptr_t)(pin + off);
        unsigned lds = (unsigned)(uintptr_t)(void*)tin;   // flat->LDS low 32 bits
        tdm_load_tile(lds, ga);
        __builtin_amdgcn_s_wait_tensorcnt(0);
    }
    if (tid < 64) sstat[tid] = 0.0f;
    __syncthreads();

    int lane = tid & 31, wave = tid >> 5;
    int mt = wave & 3, nt = wave >> 2;
    int kh = lane >> 4, lm = lane & 15;

    v8f acc = {0.f, 0.f, 0.f, 0.f, 0.f, 0.f, 0.f, 0.f};
    for (int dy = 0; dy < 7; dy++) {
        for (int dx = 0; dx < 7; dx++) {
            int t = dy * 7 + dx;
            const _Float16* ap = &tin[(((mt + dy) * 22) + (lm + dx)) * 32 + kh * 8];
            const _Float16* bp = &wt[((size_t)(t * 32) + nt * 16 + lm) * 32 + kh * 8];
            v16h a, b;
            #pragma unroll
            for (int i = 0; i < 8; i++) {
                a[i]     = ap[i];
                a[8 + i] = ap[16 + i];
                b[i]     = bp[i];
                b[8 + i] = bp[16 + i];
            }
            acc = __builtin_amdgcn_wmma_f32_16x16x32_f16(
                false, a, false, b, (short)0, acc, false, false);
        }
    }

    int n = nt * 16 + lm;
    float bsv = bias[n];
    float s1 = 0.0f, s2 = 0.0f;
    int oy = y0 + mt;
    #pragma unroll
    for (int r = 0; r < 8; r++) {
        int m = r + kh * 8;
        float v = acc[r] + bsv;
        out[(((size_t)img * 32 + n) * WDIM + oy) * WDIM + (x0 + m)] = v;
        s1 += v; s2 += v * v;
    }
    atomicAdd(&sstat[n], s1);
    atomicAdd(&sstat[32 + n], s2);
    __syncthreads();
    if (tid < 64) atomicAdd(&stats[tid], sstat[tid]);
}

__global__ void k_bnfin(const float* __restrict__ stats,
                        const float* __restrict__ g, const float* __restrict__ b,
                        float* __restrict__ sc, float* __restrict__ sh) {
    int c = threadIdx.x; if (c >= 32) return;
    float N = (float)(NBATCH * NPIX);
    float m = stats[c] / N;
    float var = stats[32 + c] / N - m * m;
    float inv = rsqrtf(var + EPSB);
    sc[c] = g[c] * inv;
    sh[c] = b[c] - m * g[c] * inv;
}

// jc5: 3x3 conv 32->1 with bn4-affine+relu on input; per-image sums
__global__ __launch_bounds__(256) void k_jc5(const float* __restrict__ in,
                                             const float* __restrict__ sc,
                                             const float* __restrict__ sh,
                                             const float* __restrict__ w5,
                                             const float* __restrict__ b5,
                                             float* __restrict__ out,
                                             float* __restrict__ xsum) {
    __shared__ float red[256];
    int gid = blockIdx.x * 256 + threadIdx.x;        // < NBATCH*NPIX
    int b = gid >> 14, pix = gid & (NPIX - 1);
    int y = pix >> 7, x = pix & (WDIM - 1);
    float acc = b5[0];
    for (int c = 0; c < 32; c++) {
        float scc = sc[c], shc = sh[c];
        const float* inc = &in[(((size_t)b * 32 + c) * WDIM) * WDIM];
        #pragma unroll
        for (int dy = -1; dy <= 1; dy++) {
            int yy = y + dy; if (yy < 0 || yy >= WDIM) continue;
            #pragma unroll
            for (int dx = -1; dx <= 1; dx++) {
                int xx = x + dx; if (xx < 0 || xx >= WDIM) continue;
                float v = fmaxf(fmaf(scc, inc[yy * WDIM + xx], shc), 0.0f);
                acc = fmaf(w5[(c * 3 + (dy + 1)) * 3 + (dx + 1)], v, acc);
            }
        }
    }
    out[gid] = acc;
    red[threadIdx.x] = acc; __syncthreads();
    for (int st = 128; st > 0; st >>= 1) {
        if (threadIdx.x < st) red[threadIdx.x] += red[threadIdx.x + st];
        __syncthreads();
    }
    if (threadIdx.x == 0) atomicAdd(&xsum[b], red[0]);
}

__global__ __launch_bounds__(256) void k_out2(const float* __restrict__ raw5,
                                              const float* __restrict__ xsum,
                                              float* __restrict__ out) {
    int gid = blockIdx.x * 256 + threadIdx.x;
    out[gid] = raw5[gid] / xsum[gid >> 14];
}

// ---------------------------------------------------------------- launch

extern "C" void kernel_launch(void* const* d_in, const int* in_sizes, int n_in,
                              void* d_out, int out_size, void* d_ws, size_t ws_size,
                              hipStream_t stream) {
    (void)in_sizes; (void)n_in; (void)out_size; (void)ws_size;
    const float* pos  = (const float*)d_in[0];
    const float* beam = (const float*)d_in[1];
    const float* c1w = (const float*)d_in[2],  *c1b = (const float*)d_in[3];
    const float* b1g = (const float*)d_in[4],  *b1b = (const float*)d_in[5];
    const float* c6w = (const float*)d_in[6],  *c6b = (const float*)d_in[7];
    const float* b6g = (const float*)d_in[8],  *b6b = (const float*)d_in[9];
    const float* c7w = (const float*)d_in[10], *c7b = (const float*)d_in[11];
    const float* j1w = (const float*)d_in[12], *j1b = (const float*)d_in[13];
    const float* jb1g = (const float*)d_in[14], *jb1b = (const float*)d_in[15];
    const float* j2w = (const float*)d_in[16], *j2b = (const float*)d_in[17];
    const float* jb2g = (const float*)d_in[18], *jb2b = (const float*)d_in[19];
    const float* j3w = (const float*)d_in[20], *j3b = (const float*)d_in[21];
    const float* jb3g = (const float*)d_in[22], *jb3b = (const float*)d_in[23];
    const float* j4w = (const float*)d_in[24], *j4b = (const float*)d_in[25];
    const float* jb4g = (const float*)d_in[26], *jb4b = (const float*)d_in[27];
    const float* j5w = (const float*)d_in[28], *j5b = (const float*)d_in[29];
    float* outp = (float*)d_out;

    // workspace carve-out (units: floats)
    float* wsf = (float*)d_ws;
    size_t o = 0;
    auto take = [&](size_t nf) { float* p = wsf + o; o += nf; return p; };
    float* vstats = take(2);
    float* zstats = take(2 * IDIM);
    float* dsum   = take(BSEQ);
    float* xsumb  = take(NBATCH);
    float* st1 = take(64); float* st2 = take(64);
    float* st3 = take(64); float* st4 = take(64);
    int n_stats = (int)o;                           // zeroed every call
    float* invb   = take(BSEQ);
    float* alpha1 = take(32); float* beta1 = take(32);
    float* a6 = take(IDIM);   float* c6 = take(IDIM);
    float* sc1 = take(32); float* sh1 = take(32);
    float* sc2 = take(32); float* sh2 = take(32);
    float* sc3 = take(32); float* sh3 = take(32);
    float* sc4 = take(32); float* sh4 = take(32);
    o = (o + 255) & ~(size_t)255;
    float* vals = take(NVAL);
    float* xs   = take(NVAL);
    float* rawA = take((size_t)NBATCH * 32 * NPIX);
    float* rawB = take((size_t)NBATCH * 32 * NPIX);
    float* raw5 = take((size_t)NBATCH * NPIX);
    _Float16* pf  = (_Float16*)(wsf + o); o += ((size_t)PADN * 32) / 2;  // padded f16 acts
    _Float16* w1h = (_Float16*)(wsf + o); o += 49 * 32 * 32 / 2;
    _Float16* w2h = (_Float16*)(wsf + o); o += 49 * 32 * 32 / 2;
    _Float16* w3h = (_Float16*)(wsf + o); o += 49 * 32 * 32 / 2;
    _Float16* w4h = (_Float16*)(wsf + o); o += 49 * 32 * 32 / 2;

    // 0) reset accumulators (graph-replay safe)
    k_zero<<<(n_stats + 255) / 256, 256, 0, stream>>>(wsf, n_stats);

    // 1) front end: gather + folded 1x1 chain with batch-stats BN
    k_beam<<<BSEQ, 256, 0, stream>>>(beam, invb);
    k_vals<<<NVAL / 256, 256, 0, stream>>>(pos, beam, invb, vals, vstats);
    k_aff1<<<1, 32, 0, stream>>>(vstats, c1w, c1b, b1g, b1b, alpha1, beta1);
    k_zstats<<<NVAL / 256, 256, 0, stream>>>(vals, alpha1, beta1, c6w, c6b, zstats);
    k_aff6<<<1, 32, 0, stream>>>(zstats, b6g, b6b, a6, c6);
    k_xs<<<NVAL / 256, 256, 0, stream>>>(vals, alpha1, beta1, c6w, c6b, a6, c6,
                                         c7w, c7b, xs, dsum);
    k_dets<<<NVAL / 256, 256, 0, stream>>>(xs, dsum, outp);           // output 1

    // 2) pack weights; build padded f16 feature tensor
    int wblk = (49 * 32 * 32 + 255) / 256;
    k_prepw<<<wblk, 256, 0, stream>>>(j1w, 19, w1h);
    k_prepw<<<wblk, 256, 0, stream>>>(j2w, 32, w2h);
    k_prepw<<<wblk, 256, 0, stream>>>(j3w, 32, w3h);
    k_prepw<<<wblk, 256, 0, stream>>>(j4w, 32, w4h);
    int pblk = (PADN + 255) / 256;
    k_feat_pad<<<pblk, 256, 0, stream>>>(xs, pf);

    // 3) WMMA conv stack; TDM stages tiles, BN affine+ReLU folded into k_xpad
    dim3 cgrid(WDIM / 16, WDIM / 4, NBATCH);
    k_conv7<<<cgrid, 256, 0, stream>>>(pf, w1h, j1b, rawA, st1);
    k_bnfin<<<1, 32, 0, stream>>>(st1, jb1g, jb1b, sc1, sh1);
    k_xpad<<<pblk, 256, 0, stream>>>(rawA, sc1, sh1, pf);
    k_conv7<<<cgrid, 256, 0, stream>>>(pf, w2h, j2b, rawB, st2);
    k_bnfin<<<1, 32, 0, stream>>>(st2, jb2g, jb2b, sc2, sh2);
    k_xpad<<<pblk, 256, 0, stream>>>(rawB, sc2, sh2, pf);
    k_conv7<<<cgrid, 256, 0, stream>>>(pf, w3h, j3b, rawA, st3);
    k_bnfin<<<1, 32, 0, stream>>>(st3, jb3g, jb3b, sc3, sh3);
    k_xpad<<<pblk, 256, 0, stream>>>(rawA, sc3, sh3, pf);
    k_conv7<<<cgrid, 256, 0, stream>>>(pf, w4h, j4b, rawB, st4);
    k_bnfin<<<1, 32, 0, stream>>>(st4, jb4g, jb4b, sc4, sh4);

    // 4) jc5 (3x3, 32->1) + final normalization
    k_jc5<<<NBATCH * NPIX / 256, 256, 0, stream>>>(rawB, sc4, sh4, j5w, j5b, raw5, xsumb);
    k_out2<<<NBATCH * NPIX / 256, 256, 0, stream>>>(raw5, xsumb, outp + NVAL); // output 2
}
